// DrugGraphEmbedding_11836929868222
// MI455X (gfx1250) — compile-verified, matrix operand order checked by reading, not run on hardware
//
#include <hip/hip_runtime.h>

typedef __attribute__((ext_vector_type(16))) _Float16 v16h;
typedef __attribute__((ext_vector_type(8)))  float    v8f;
typedef __attribute__((ext_vector_type(2)))  _Float16 h2v;
typedef __attribute__((ext_vector_type(4)))  _Float16 h4v;

#define N_NODES  200000
#define N_EDGES  3200000
#define N_GRAPHS 4096
#define IN_CH    78
#define H1_DIM   64
#define H2_DIM   128
#define OUT_DIM  128

// ---------------------------------------------------------------- utilities
__global__ void fill_kernel(float* __restrict__ p, float v, long n) {
  long i = (long)blockIdx.x * blockDim.x + threadIdx.x;
  long stride = (long)gridDim.x * blockDim.x;
  for (; i < n; i += stride) p[i] = v;
}

__global__ void deg_edges_kernel(const int* __restrict__ dst, float* __restrict__ deg, int E) {
  int e = blockIdx.x * blockDim.x + threadIdx.x;
  if (e < E) atomicAdd(&deg[dst[e]], 1.0f);
}

__global__ void rsqrt_kernel(float* __restrict__ deg, int n) {
  int i = blockIdx.x * blockDim.x + threadIdx.x;
  if (i < n) deg[i] = rsqrtf(deg[i]);   // deg >= 1 always (self loops)
}

__global__ void counts_kernel(const int* __restrict__ batch, float* __restrict__ counts, int n) {
  int i = blockIdx.x * blockDim.x + threadIdx.x;
  if (i < n) atomicAdd(&counts[batch[i]], 1.0f);
}

// ---------------------------------------------------------------- GEMM 1: hw1h = f16(x @ W1)   [200000x78 @ 78x64]
__global__ void __launch_bounds__(256) gemm1_kernel(const float* __restrict__ x,
                                                    const float* __restrict__ W,
                                                    _Float16* __restrict__ out) {
  const int NT = H1_DIM / 16;                       // 4 col tiles
  int wave = blockIdx.x * (blockDim.x >> 5) + (threadIdx.x >> 5);
  int lane = threadIdx.x & 31;
  int tm = wave / NT, tn = wave % NT;
  if (tm >= N_NODES / 16) return;                   // wave-uniform
  int row = tm * 16 + (lane & 15);
  int col = tn * 16 + (lane & 15);
  int klo = (lane >> 4) << 3;                       // 0 or 8
  v8f acc = {};
#pragma unroll
  for (int ks = 0; ks < 3; ++ks) {                  // K padded 78 -> 96
    v16h a, b;
#pragma unroll
    for (int i = 0; i < 16; ++i) {
      int k = ks * 32 + ((i >> 3) << 4) + klo + (i & 7);
      // Branch-free zero padding: load from a clamped (always in-bounds)
      // index and select 0 for the padded lanes -> v_cndmask, no exec churn.
      int kc = (k < IN_CH) ? k : (IN_CH - 1);
      float av = x[(long)row * IN_CH + kc];
      float bv = W[(long)kc * H1_DIM + col];
      float m  = (k < IN_CH) ? 1.0f : 0.0f;
      a[i] = (_Float16)(av * m);
      b[i] = (_Float16)(bv * m);
    }
    acc = __builtin_amdgcn_wmma_f32_16x16x32_f16(false, a, false, b, (short)0, acc, false, false);
  }
  int mbase = tm * 16 + ((lane >> 4) << 3);
#pragma unroll
  for (int j = 0; j < 8; ++j)
    out[(long)(mbase + j) * H1_DIM + col] = (_Float16)acc[j];
}

// ---------------------------------------------------------------- GEMM 2: hw2h = f16(relu(agg1 + b1) @ W2)   [200000x64 @ 64x128]
__global__ void __launch_bounds__(256) gemm2_kernel(const float* __restrict__ agg1,
                                                    const float* __restrict__ b1,
                                                    const float* __restrict__ W,
                                                    _Float16* __restrict__ out) {
  const int NT = H2_DIM / 16;                       // 8 col tiles
  int wave = blockIdx.x * (blockDim.x >> 5) + (threadIdx.x >> 5);
  int lane = threadIdx.x & 31;
  int tm = wave / NT, tn = wave % NT;
  if (tm >= N_NODES / 16) return;
  int row = tm * 16 + (lane & 15);
  int col = tn * 16 + (lane & 15);
  int klo = (lane >> 4) << 3;
  v8f acc = {};
#pragma unroll
  for (int ks = 0; ks < 2; ++ks) {                  // K = 64 exact
    v16h a, b;
#pragma unroll
    for (int i = 0; i < 16; ++i) {
      int k = ks * 32 + ((i >> 3) << 4) + klo + (i & 7);
      float av = fmaxf(agg1[(long)row * H1_DIM + k] + b1[k], 0.0f);  // fused bias+relu
      a[i] = (_Float16)av;
      b[i] = (_Float16)W[(long)k * H2_DIM + col];
    }
    acc = __builtin_amdgcn_wmma_f32_16x16x32_f16(false, a, false, b, (short)0, acc, false, false);
  }
  int mbase = tm * 16 + ((lane >> 4) << 3);
#pragma unroll
  for (int j = 0; j < 8; ++j)
    out[(long)(mbase + j) * H2_DIM + col] = (_Float16)acc[j];
}

// ---------------------------------------------------------------- edge scatter: agg[dst] += dinv[s]*dinv[d] * msg[src]
template <int F>
__global__ void __launch_bounds__(256) scatter_kernel(const int* __restrict__ src,
                                                      const int* __restrict__ dst,
                                                      const float* __restrict__ dinv,
                                                      const _Float16* __restrict__ msg,
                                                      float* __restrict__ agg) {
  constexpr int V = F / 32;                         // halves per lane (2 or 4)
  long wave = (long)blockIdx.x * (blockDim.x >> 5) + (threadIdx.x >> 5);
  int lane = threadIdx.x & 31;
  const long total = (long)N_EDGES + N_NODES;       // real edges + self loops
  if (wave >= total) return;
  int s, d;
  if (wave < N_EDGES) { s = src[wave]; d = dst[wave]; }
  else                { s = d = (int)(wave - N_EDGES); }
  float w = dinv[s] * dinv[d];
  const _Float16* ps = msg + (long)s * F + lane * V;
  float*          pd = agg + (long)d * F + lane * V;
  if constexpr (V == 2) {
    h2v v = *(const h2v*)ps;
    atomicAdd(pd + 0, w * (float)v[0]);
    atomicAdd(pd + 1, w * (float)v[1]);
  } else {
    h4v v = *(const h4v*)ps;
#pragma unroll
    for (int i = 0; i < 4; ++i) atomicAdd(pd + i, w * (float)v[i]);
  }
}

// ---------------------------------------------------------------- sorted-segment mean-pool accumulate: pooled[g] += relu(agg2 + b2)
__global__ void __launch_bounds__(256) pool_kernel(const float* __restrict__ agg2,
                                                   const float* __restrict__ b2,
                                                   const int* __restrict__ batch,
                                                   float* __restrict__ pooled) {
  int lane = threadIdx.x & 31;
  long wave = (long)blockIdx.x * (blockDim.x >> 5) + (threadIdx.x >> 5);
  long n0 = wave * 32;
  if (n0 >= N_NODES) return;
  long n1 = n0 + 32; if (n1 > N_NODES) n1 = N_NODES;
  float4 bias = ((const float4*)b2)[lane];
  float4 acc = make_float4(0.f, 0.f, 0.f, 0.f);
  int curg = batch[n0];
  for (long n = n0; n < n1; ++n) {
    int g = batch[n];
    if (g != curg) {
      float* p = pooled + (long)curg * H2_DIM + lane * 4;
      atomicAdd(p + 0, acc.x); atomicAdd(p + 1, acc.y);
      atomicAdd(p + 2, acc.z); atomicAdd(p + 3, acc.w);
      acc = make_float4(0.f, 0.f, 0.f, 0.f);
      curg = g;
    }
    float4 v = ((const float4*)(agg2 + (long)n * H2_DIM))[lane];
    acc.x += fmaxf(v.x + bias.x, 0.0f);
    acc.y += fmaxf(v.y + bias.y, 0.0f);
    acc.z += fmaxf(v.z + bias.z, 0.0f);
    acc.w += fmaxf(v.w + bias.w, 0.0f);
  }
  float* p = pooled + (long)curg * H2_DIM + lane * 4;
  atomicAdd(p + 0, acc.x); atomicAdd(p + 1, acc.y);
  atomicAdd(p + 2, acc.z); atomicAdd(p + 3, acc.w);
}

// ---------------------------------------------------------------- FC: out = (pooled/count) @ fc_w + fc_b   [4096x128 @ 128x128]
__global__ void __launch_bounds__(256) fc_kernel(const float* __restrict__ pooled,
                                                 const float* __restrict__ counts,
                                                 const float* __restrict__ fcw,
                                                 const float* __restrict__ fcb,
                                                 float* __restrict__ out) {
  const int NT = OUT_DIM / 16;                      // 8
  int wave = blockIdx.x * (blockDim.x >> 5) + (threadIdx.x >> 5);
  int lane = threadIdx.x & 31;
  int tm = wave / NT, tn = wave % NT;
  if (tm >= N_GRAPHS / 16) return;
  int row = tm * 16 + (lane & 15);
  int col = tn * 16 + (lane & 15);
  int klo = (lane >> 4) << 3;
  float scale = 1.0f / fmaxf(counts[row], 1.0f);    // fused mean division
  v8f acc = {};
#pragma unroll
  for (int ks = 0; ks < 4; ++ks) {                  // K = 128 exact
    v16h a, b;
#pragma unroll
    for (int i = 0; i < 16; ++i) {
      int k = ks * 32 + ((i >> 3) << 4) + klo + (i & 7);
      a[i] = (_Float16)(pooled[(long)row * H2_DIM + k] * scale);
      b[i] = (_Float16)fcw[(long)k * OUT_DIM + col];
    }
    acc = __builtin_amdgcn_wmma_f32_16x16x32_f16(false, a, false, b, (short)0, acc, false, false);
  }
  int mbase = tm * 16 + ((lane >> 4) << 3);
  float bias = fcb[col];
#pragma unroll
  for (int j = 0; j < 8; ++j)
    out[(long)(mbase + j) * OUT_DIM + col] = acc[j] + bias;
}

// ---------------------------------------------------------------- host
extern "C" void kernel_launch(void* const* d_in, const int* in_sizes, int n_in,
                              void* d_out, int out_size, void* d_ws, size_t ws_size,
                              hipStream_t stream) {
  (void)in_sizes; (void)n_in; (void)out_size; (void)ws_size;
  const float* x     = (const float*)d_in[0];
  const int*   ei    = (const int*)d_in[1];   // [2, E] : row 0 = src, row 1 = dst
  const int*   batch = (const int*)d_in[2];
  const float* W1    = (const float*)d_in[3];
  const float* b1    = (const float*)d_in[4];
  const float* W2    = (const float*)d_in[5];
  const float* b2    = (const float*)d_in[6];
  const float* fcw   = (const float*)d_in[7];
  const float* fcb   = (const float*)d_in[8];
  float* out = (float*)d_out;
  const int* src = ei;
  const int* dst = ei + N_EDGES;

  // workspace carve-out (256B aligned): ~233 MB total
  char* ws = (char*)d_ws;
  auto carve = [&](size_t bytes) {
    char* p = ws;
    ws += (bytes + 255) & ~(size_t)255;
    return (void*)p;
  };
  float*    dinv   = (float*)   carve(sizeof(float)    * (size_t)N_NODES);
  _Float16* hw1h   = (_Float16*)carve(sizeof(_Float16) * (size_t)N_NODES * H1_DIM);
  float*    agg1   = (float*)   carve(sizeof(float)    * (size_t)N_NODES * H1_DIM);
  _Float16* hw2h   = (_Float16*)carve(sizeof(_Float16) * (size_t)N_NODES * H2_DIM);
  float*    agg2   = (float*)   carve(sizeof(float)    * (size_t)N_NODES * H2_DIM);
  float*    pooled = (float*)   carve(sizeof(float)    * (size_t)N_GRAPHS * H2_DIM);
  float*    counts = (float*)   carve(sizeof(float)    * (size_t)N_GRAPHS);

  // zero/init accumulators (deg starts at 1.0 for the self loop)
  fill_kernel<<<1024, 256, 0, stream>>>(dinv, 1.0f, (long)N_NODES);
  fill_kernel<<<2048, 256, 0, stream>>>(agg1, 0.0f, (long)N_NODES * H1_DIM);
  fill_kernel<<<4096, 256, 0, stream>>>(agg2, 0.0f, (long)N_NODES * H2_DIM);
  fill_kernel<<<512,  256, 0, stream>>>(pooled, 0.0f, (long)N_GRAPHS * H2_DIM);
  fill_kernel<<<16,   256, 0, stream>>>(counts, 0.0f, (long)N_GRAPHS);

  // degree -> dinv
  deg_edges_kernel<<<(N_EDGES + 255) / 256, 256, 0, stream>>>(dst, dinv, N_EDGES);
  rsqrt_kernel<<<(N_NODES + 255) / 256, 256, 0, stream>>>(dinv, N_NODES);

  // layer 1
  { int waves = (N_NODES / 16) * (H1_DIM / 16);          // 50000
    gemm1_kernel<<<(waves + 7) / 8, 256, 0, stream>>>(x, W1, hw1h); }
  { long waves = (long)N_EDGES + N_NODES;                // 3.4M
    scatter_kernel<H1_DIM><<<(unsigned)((waves + 7) / 8), 256, 0, stream>>>(src, dst, dinv, hw1h, agg1); }

  // layer 2 (bias+relu fused into gemm2 A-load)
  { int waves = (N_NODES / 16) * (H2_DIM / 16);          // 100000
    gemm2_kernel<<<(waves + 7) / 8, 256, 0, stream>>>(agg1, b1, W2, hw2h); }
  { long waves = (long)N_EDGES + N_NODES;
    scatter_kernel<H2_DIM><<<(unsigned)((waves + 7) / 8), 256, 0, stream>>>(src, dst, dinv, hw2h, agg2); }

  // mean pool (bias+relu fused) + FC
  counts_kernel<<<(N_NODES + 255) / 256, 256, 0, stream>>>(batch, counts, N_NODES);
  { long waves = (N_NODES + 31) / 32;                    // 6250
    pool_kernel<<<(unsigned)((waves + 7) / 8), 256, 0, stream>>>(agg2, b2, batch, pooled); }
  { int waves = (N_GRAPHS / 16) * (OUT_DIM / 16);        // 2048
    fc_kernel<<<(waves + 7) / 8, 256, 0, stream>>>(pooled, counts, fcw, fcb, out); }
}
